// AttentionNet_40166534152918
// MI455X (gfx1250) — matrix operand order repaired and checked by reading, not compiled, not run on hardware
//
#include <hip/hip_runtime.h>
#include <cstdint>
#include <cstddef>

// ---------------------------------------------------------------------------
// Types / helpers
// ---------------------------------------------------------------------------
typedef __bf16 bf16;
typedef __attribute__((ext_vector_type(16))) __bf16 v16bf;
typedef __attribute__((ext_vector_type(8)))  __bf16 v8bf;
typedef __attribute__((ext_vector_type(8)))  float  v8f;

#define DEV_INLINE __device__ __forceinline__

DEV_INLINE float bf2f(bf16 x) {
  unsigned short u = __builtin_bit_cast(unsigned short, x);
  unsigned int w = ((unsigned int)u) << 16;
  return __builtin_bit_cast(float, w);
}
DEV_INLINE bf16 f2bf(float f) {
  unsigned int u = __builtin_bit_cast(unsigned int, f);
  u += 0x7FFFu + ((u >> 16) & 1u);               // round-to-nearest-even
  return __builtin_bit_cast(bf16, (unsigned short)(u >> 16));
}

// A fragment (16x32 bf16, row-major source). lane: r = lane&15 (row), half = lane>>4.
// e[i] = A[r, kh+i] for i<16, kh = half*16 (contiguous k order).
DEV_INLINE v16bf a_frag(const bf16* rowk0, int half) {
  v8bf lo = *(const v8bf*)(rowk0 + half * 16);
  v8bf hi = *(const v8bf*)(rowk0 + half * 16 + 8);
  return __builtin_shufflevector(lo, hi, 0, 1, 2, 3, 4, 5, 6, 7,
                                 8, 9, 10, 11, 12, 13, 14, 15);
}

DEV_INLINE v8f wmma_bf16(v16bf a, v16bf b, v8f c) {
  return __builtin_amdgcn_wmma_f32_16x16x32_bf16(false, a, false, b, (short)0, c,
                                                 false, false);
}

// ---------------------------------------------------------------------------
// Weight prep: all weights stored TRANSPOSED as WT[N,K] bf16 so that a B
// fragment (e[i] = B[kh+i, n] = WT[n, kh+i]) is one contiguous 32B load.
// ---------------------------------------------------------------------------
// generic [K,N] fp32 -> [N,K] bf16
__global__ void castT_kernel(const float* __restrict__ s, bf16* __restrict__ d,
                             int K, int N) {
  int i = blockIdx.x * blockDim.x + threadIdx.x;
  if (i >= K * N) return;
  int k = i / N, n = i % N;
  d[(size_t)n * K + k] = f2bf(s[i]);
}
// [H,E,D] (4,128,32) -> WT[(h*32+d), e]  (i.e. [N=128, K=128])
__global__ void pack_qkvT_kernel(const float* __restrict__ w, bf16* __restrict__ o) {
  int i = blockIdx.x * blockDim.x + threadIdx.x;
  if (i >= 4 * 128 * 32) return;
  int h = i / (128 * 32), e = (i >> 5) & 127, d = i & 31;
  o[(size_t)(h * 32 + d) * 128 + e] = f2bf(w[i]);
}
__global__ void cast_kernel(const float* __restrict__ s, bf16* __restrict__ d, int n) {
  int i = blockIdx.x * blockDim.x + threadIdx.x;
  if (i < n) d[i] = f2bf(s[i]);
}

// ---------------------------------------------------------------------------
// LayerNorm over E=128, one wave per row -> bf16
// ---------------------------------------------------------------------------
__global__ void layernorm_kernel(const float* __restrict__ x, const float* __restrict__ g,
                                 const float* __restrict__ bta, bf16* __restrict__ y) {
  const int row = blockIdx.x, lane = threadIdx.x;
  const float4 v = ((const float4*)(x + (size_t)row * 128))[lane];
  float s = v.x + v.y + v.z + v.w;
#pragma unroll
  for (int o = 16; o > 0; o >>= 1) s += __shfl_xor(s, o);
  const float mean = s * (1.0f / 128.0f);
  const float dx = v.x - mean, dy = v.y - mean, dz = v.z - mean, dw = v.w - mean;
  float q = dx * dx + dy * dy + dz * dz + dw * dw;
#pragma unroll
  for (int o = 16; o > 0; o >>= 1) q += __shfl_xor(q, o);
  const float rs = rsqrtf(q * (1.0f / 128.0f) + 1e-5f);
  const float4 gg = ((const float4*)g)[lane];
  const float4 bb = ((const float4*)bta)[lane];
  bf16* yp = y + (size_t)row * 128 + lane * 4;
  yp[0] = f2bf(dx * rs * gg.x + bb.x);
  yp[1] = f2bf(dy * rs * gg.y + bb.y);
  yp[2] = f2bf(dz * rs * gg.z + bb.z);
  yp[3] = f2bf(dw * rs * gg.w + bb.w);
}

// ---------------------------------------------------------------------------
// WMMA GEMM: C[M,N] = act(A[M,K] * W[K,N] + bias) + resid.
// W passed transposed: WT[N,K] (ldwt = K). One wave per 16(M) x 64(N) block:
// 4 n-tiles share each A fragment (4 independent WMMAs per k-step).
// Optional transposed bf16 output Ct[(row/1024)*ldc + col][row%1024] (for V).
// ---------------------------------------------------------------------------
__global__ void gemm_wmma_kernel(const bf16* __restrict__ A, int lda,
                                 const bf16* __restrict__ WT, int ldwt,
                                 float* __restrict__ Cf, bf16* __restrict__ Cb,
                                 bf16* __restrict__ Ct, int ldc,
                                 int K, const float* __restrict__ bias,
                                 const float* __restrict__ resid, int relu) {
  const int tn0 = blockIdx.x * 4, tm = blockIdx.y;
  const int lane = threadIdx.x;
  const int r = lane & 15, half = lane >> 4;
  const bf16* Arow = A + (size_t)(tm * 16 + r) * lda;
  v8f acc[4];
#pragma unroll
  for (int j = 0; j < 4; ++j) acc[j] = (v8f){0.f, 0.f, 0.f, 0.f, 0.f, 0.f, 0.f, 0.f};
  for (int k0 = 0; k0 < K; k0 += 32) {
    __builtin_prefetch(Arow + k0 + 128, 0, 3);
    v16bf a = a_frag(Arow + k0, half);
#pragma unroll
    for (int j = 0; j < 4; ++j) {
      const bf16* bp = WT + (size_t)((tn0 + j) * 16 + r) * ldwt + k0 + half * 16;
      v16bf b = *(const v16bf*)bp;   // 32B contiguous: e[i] = WT[n, k0+kh+i]
      acc[j] = wmma_bf16(a, b, acc[j]);
    }
  }
#pragma unroll
  for (int j = 0; j < 4; ++j) {
    const int col = (tn0 + j) * 16 + r;
#pragma unroll
    for (int vr = 0; vr < 8; ++vr) {
      const size_t row = (size_t)tm * 16 + half * 8 + vr;
      float v = acc[j][vr];
      if (bias) v += bias[col];
      if (relu) v = fmaxf(v, 0.0f);
      if (resid) v += resid[row * ldc + col];
      if (Cf) Cf[row * ldc + col] = v;
      if (Cb) Cb[row * ldc + col] = f2bf(v);
      if (Ct) Ct[((row >> 10) * (size_t)ldc + col) * 1024 + (row & 1023)] = f2bf(v);
    }
  }
}

// ---------------------------------------------------------------------------
// Encoder attention: one wave per (b, h, 16-query tile). fp32 scores in LDS
// (16 x 1024 = 64 KB); softmax is register-resident (32 vals/lane/row) and
// emits packed bf16 probability rows IN PLACE at (bf16*)S + row*1024 (row r's
// bf16 row only overlays fp32 row r/2 <= r, already consumed). P*V then uses
// contiguous ds_load_b128 A fragments and contiguous Vt B fragments.
// ---------------------------------------------------------------------------
__global__ void enc_attn_kernel(const bf16* __restrict__ Q, const bf16* __restrict__ K,
                                const bf16* __restrict__ Vt, const int* __restrict__ mask,
                                bf16* __restrict__ outp) {
  extern __shared__ float S[];  // [16][1024] fp32 scores; later [16][1024] bf16 probs
  const int qt = blockIdx.x, h = blockIdx.y, b = blockIdx.z;
  const int lane = threadIdx.x;
  const int r = lane & 15, half = lane >> 4;
  const float scale = 0.1767766952966369f;  // 1/sqrt(32)

  const bf16* Qbase = Q + ((size_t)b * 1024 + qt * 16) * 128 + h * 32;
  v16bf aq = a_frag(Qbase + (size_t)r * 128, half);

  // S = Q * K^T   (B fragment: e[i] = K[t0+n, kh+i], contiguous in d)
  for (int t0 = 0; t0 < 1024; t0 += 16) {
    const bf16* kp = K + ((size_t)b * 1024 + t0 + r) * 128 + h * 32 + half * 16;
    v16bf bk = *(const v16bf*)kp;
    v8f sacc = {0.f, 0.f, 0.f, 0.f, 0.f, 0.f, 0.f, 0.f};
    sacc = wmma_bf16(aq, bk, sacc);
#pragma unroll
    for (int vr = 0; vr < 8; ++vr)
      S[(half * 8 + vr) * 1024 + t0 + r] = sacc[vr];
  }
  __syncthreads();

  // masked softmax, register-resident per row; write bf16 probs in place
  bf16* P = (bf16*)S;
  for (int row = 0; row < 16; ++row) {
    const int* mrow = mask + ((size_t)b * 1024 + (qt * 16 + row)) * 1024;
    const float* srow = S + row * 1024;
    bf16* prow = P + row * 1024;
    float vals[32];
    float mx = -__builtin_inff();
#pragma unroll
    for (int j = 0; j < 32; ++j) {
      const int t = j * 32 + lane;
      float v = srow[t] * scale;
      if (mrow[t]) v = -__builtin_inff();
      vals[j] = v;
      mx = fmaxf(mx, v);
    }
#pragma unroll
    for (int o = 16; o > 0; o >>= 1) mx = fmaxf(mx, __shfl_xor(mx, o));
    float sum = 0.0f;
#pragma unroll
    for (int j = 0; j < 32; ++j) {
      float e = (vals[j] == -__builtin_inff()) ? 0.0f : __expf(vals[j] - mx);
      vals[j] = e;
      sum += e;
    }
#pragma unroll
    for (int o = 16; o > 0; o >>= 1) sum += __shfl_xor(sum, o);
    const float inv = 1.0f / sum;
#pragma unroll
    for (int j = 0; j < 32; ++j) prow[j * 32 + lane] = f2bf(vals[j] * inv);
  }
  __syncthreads();

  // O = P * V: A fragment from packed bf16 LDS rows (two ds_load_b128);
  // B fragment e[i] = V[t0+kh+i, n] = Vt[b, n, t0+kh+i] (contiguous 32B).
  v8f o0 = {0.f, 0.f, 0.f, 0.f, 0.f, 0.f, 0.f, 0.f};
  v8f o1 = {0.f, 0.f, 0.f, 0.f, 0.f, 0.f, 0.f, 0.f};
  const bf16* vbase = Vt + ((size_t)b * 128 + h * 32 + r) * 1024;
  for (int t0 = 0; t0 < 1024; t0 += 32) {
    v16bf ap = a_frag(P + r * 1024 + t0, half);
    v16bf bv0 = *(const v16bf*)(vbase + t0 + half * 16);
    v16bf bv1 = *(const v16bf*)(vbase + 16 * 1024 + t0 + half * 16);
    o0 = wmma_bf16(ap, bv0, o0);
    o1 = wmma_bf16(ap, bv1, o1);
  }
  bf16* op = outp + ((size_t)b * 1024 + qt * 16) * 128 + h * 32;
#pragma unroll
  for (int vr = 0; vr < 8; ++vr) {
    const int row = half * 8 + vr;
    op[(size_t)row * 128 + r] = f2bf(o0[vr]);
    op[(size_t)row * 128 + 16 + r] = f2bf(o1[vr]);
  }
}

// ---------------------------------------------------------------------------
// Small matmul (decoder-side, M<=32): C = act(A_bf16 * WT_bf16 + bias) + resid
// WT is [N,K] row-major (ldw = K) -> K-loop is contiguous for both operands.
// ---------------------------------------------------------------------------
__global__ void small_mm_kernel(const bf16* __restrict__ A, int lda,
                                const bf16* __restrict__ WT, int ldw,
                                float* __restrict__ Cf, bf16* __restrict__ Cb, int ldc,
                                int M, int N, int K,
                                const float* __restrict__ bias,
                                const float* __restrict__ resid, int relu) {
  int idx = blockIdx.x * blockDim.x + threadIdx.x;
  if (idx >= M * N) return;
  int m = idx / N, n = idx % N;
  float acc = 0.0f;
  const bf16* ap = A + (size_t)m * lda;
  const bf16* wp = WT + (size_t)n * ldw;
  for (int k = 0; k < K; ++k) acc += bf2f(ap[k]) * bf2f(wp[k]);
  if (bias) acc += bias[n];
  if (relu) acc = fmaxf(acc, 0.0f);
  if (resid) acc += resid[(size_t)m * ldc + n];
  if (Cf) Cf[(size_t)m * ldc + n] = acc;
  if (Cb) Cb[(size_t)m * ldc + n] = f2bf(acc);
}

// ---------------------------------------------------------------------------
// Decoder cross-attention, Q=1 per batch. One block per (b, h). V transposed.
// ---------------------------------------------------------------------------
__global__ void dec_attn_kernel(const float* __restrict__ qd, const bf16* __restrict__ K,
                                const bf16* __restrict__ Vt, const int* __restrict__ mask,
                                bf16* __restrict__ heads) {
  const int b = blockIdx.x, h = blockIdx.y, tid = threadIdx.x;  // 256 threads
  __shared__ float sc[1024];
  __shared__ float rbuf[256];
  __shared__ float ob[8][32];
  float qv[32];
  const float* qrow = qd + b * 128 + h * 32;
#pragma unroll
  for (int d = 0; d < 32; ++d) qv[d] = qrow[d];
  const float scale = 0.1767766952966369f;
  float mx = -__builtin_inff();
  for (int t = tid; t < 1024; t += 256) {
    const bf16* kp = K + ((size_t)b * 1024 + t) * 128 + h * 32;
    float acc = 0.0f;
#pragma unroll
    for (int d = 0; d < 32; ++d) acc += qv[d] * bf2f(kp[d]);
    acc *= scale;
    if (mask[(size_t)b * 1024 + t]) acc = -__builtin_inff();
    sc[t] = acc;
    mx = fmaxf(mx, acc);
  }
  rbuf[tid] = mx; __syncthreads();
  for (int s2 = 128; s2 > 0; s2 >>= 1) {
    if (tid < s2) rbuf[tid] = fmaxf(rbuf[tid], rbuf[tid + s2]);
    __syncthreads();
  }
  mx = rbuf[0]; __syncthreads();
  float sum = 0.0f;
  for (int t = tid; t < 1024; t += 256) {
    float v = sc[t];
    float e = (v == -__builtin_inff()) ? 0.0f : __expf(v - mx);
    sc[t] = e;
    sum += e;
  }
  rbuf[tid] = sum; __syncthreads();
  for (int s2 = 128; s2 > 0; s2 >>= 1) {
    if (tid < s2) rbuf[tid] += rbuf[tid + s2];
    __syncthreads();
  }
  const float inv = 1.0f / rbuf[0];
  const int d = tid & 31, part = tid >> 5;
  const bf16* vrow = Vt + ((size_t)b * 128 + h * 32 + d) * 1024;  // contiguous in t
  float acc = 0.0f;
  for (int t = part * 128; t < part * 128 + 128; ++t)
    acc += sc[t] * bf2f(vrow[t]);
  ob[part][d] = acc;
  __syncthreads();
  if (tid < 32) {
    float tot = 0.0f;
#pragma unroll
    for (int p = 0; p < 8; ++p) tot += ob[p][tid];
    heads[b * 128 + h * 32 + tid] = f2bf(tot * inv);
  }
}

// ---------------------------------------------------------------------------
// Pointer head: U = 10*tanh(Qp.Kp / sqrt(E)), mask -> -1e4, log_softmax over T.
// ---------------------------------------------------------------------------
__global__ void pointer_kernel(const float* __restrict__ Qp, const bf16* __restrict__ Kp,
                               const int* __restrict__ mask, float* __restrict__ out) {
  const int b = blockIdx.x, tid = threadIdx.x;  // 256 threads
  __shared__ float qv[128];
  __shared__ float u[1024];
  __shared__ float rbuf[256];
  if (tid < 128) qv[tid] = Qp[b * 128 + tid];
  __syncthreads();
  const float scale = 0.08838834764831845f;  // 1/sqrt(128)
  float mx = -__builtin_inff();
  for (int t = tid; t < 1024; t += 256) {
    const bf16* kp = Kp + ((size_t)b * 1024 + t) * 128;
    float acc = 0.0f;
    for (int e = 0; e < 128; ++e) acc += qv[e] * bf2f(kp[e]);
    acc = 10.0f * tanhf(acc * scale);
    if (mask[(size_t)b * 1024 + t]) acc = -10000.0f;
    u[t] = acc;
    mx = fmaxf(mx, acc);
  }
  rbuf[tid] = mx; __syncthreads();
  for (int s2 = 128; s2 > 0; s2 >>= 1) {
    if (tid < s2) rbuf[tid] = fmaxf(rbuf[tid], rbuf[tid + s2]);
    __syncthreads();
  }
  mx = rbuf[0]; __syncthreads();
  float sum = 0.0f;
  for (int t = tid; t < 1024; t += 256) sum += __expf(u[t] - mx);
  rbuf[tid] = sum; __syncthreads();
  for (int s2 = 128; s2 > 0; s2 >>= 1) {
    if (tid < s2) rbuf[tid] += rbuf[tid + s2];
    __syncthreads();
  }
  const float lse = mx + logf(rbuf[0]);
  for (int t = tid; t < 1024; t += 256)
    out[(size_t)b * 1024 + t] = u[t] - lse;
}

// ---------------------------------------------------------------------------
// Host orchestration
// ---------------------------------------------------------------------------
extern "C" void kernel_launch(void* const* d_in, const int* in_sizes, int n_in,
                              void* d_out, int out_size, void* d_ws, size_t ws_size,
                              hipStream_t stream) {
  (void)in_sizes; (void)n_in; (void)out_size; (void)ws_size;
  const int Mtok = 32 * 1024;  // B*T

  const float* src       = (const float*)d_in[0];
  const float* tgt       = (const float*)d_in[1];
  const int*   enc_mask  = (const int*)d_in[2];
  const int*   dec_mask  = (const int*)d_in[3];
  const float* enc_wq    = (const float*)d_in[4];
  const float* enc_wk    = (const float*)d_in[5];
  const float* enc_wv    = (const float*)d_in[6];
  const float* enc_wo    = (const float*)d_in[7];
  const float* enc_ln1_g = (const float*)d_in[8];
  const float* enc_ln1_b = (const float*)d_in[9];
  const float* enc_ln2_g = (const float*)d_in[10];
  const float* enc_ln2_b = (const float*)d_in[11];
  const float* enc_w1    = (const float*)d_in[12];
  const float* enc_b1    = (const float*)d_in[13];
  const float* enc_w2    = (const float*)d_in[14];
  const float* enc_b2    = (const float*)d_in[15];
  const float* dec_wq    = (const float*)d_in[16];
  const float* dec_wk    = (const float*)d_in[17];
  const float* dec_wv    = (const float*)d_in[18];
  const float* dec_wo    = (const float*)d_in[19];
  const float* dec_ln1_g = (const float*)d_in[20];
  const float* dec_ln1_b = (const float*)d_in[21];
  const float* dec_ln2_g = (const float*)d_in[22];
  const float* dec_ln2_b = (const float*)d_in[23];
  const float* dec_w1    = (const float*)d_in[24];
  const float* dec_b1    = (const float*)d_in[25];
  const float* dec_w2    = (const float*)d_in[26];
  const float* dec_b2    = (const float*)d_in[27];
  const float* ptr_wq    = (const float*)d_in[28];
  const float* ptr_wk    = (const float*)d_in[29];

  char* base = (char*)d_ws;
  size_t off = 0;
  auto take = [&](size_t bytes) -> void* {
    void* p = base + off;
    off = (off + bytes + 255) & ~(size_t)255;
    return p;
  };

  // Big activation buffers (reused across phases)
  bf16*  hn   = (bf16*)take((size_t)Mtok * 128 * 2);  // enc ln1 out; reused: decoder mem_ln
  bf16*  Qh   = (bf16*)take((size_t)Mtok * 128 * 2);
  bf16*  Kh   = (bf16*)take((size_t)Mtok * 128 * 2);  // reused: decoder Kh
  bf16*  Vh   = (bf16*)take((size_t)Mtok * 128 * 2);  // transposed layout; reused: dec Vt
  bf16*  attn = (bf16*)take((size_t)Mtok * 128 * 2);  // attn out; reused: pointer Kp
  float* h1   = (float*)take((size_t)Mtok * 128 * 4);
  bf16*  ln2b = (bf16*)take((size_t)Mtok * 128 * 2);  // enc ln2 out; reused: bf16(memory)
  bf16*  mid  = (bf16*)take((size_t)Mtok * 512 * 2);
  float* mem  = (float*)take((size_t)Mtok * 128 * 4);

  // bf16 transposed weights WT[N,K]
  bf16* wqe = (bf16*)take(16384 * 2);
  bf16* wke = (bf16*)take(16384 * 2);
  bf16* wve = (bf16*)take(16384 * 2);
  bf16* woe = (bf16*)take(16384 * 2);
  bf16* w1e = (bf16*)take(65536 * 2);
  bf16* w2e = (bf16*)take(65536 * 2);
  bf16* wqd = (bf16*)take(16384 * 2);
  bf16* wkd = (bf16*)take(16384 * 2);
  bf16* wvd = (bf16*)take(16384 * 2);
  bf16* wod = (bf16*)take(16384 * 2);
  bf16* w1d = (bf16*)take(65536 * 2);
  bf16* w2d = (bf16*)take(65536 * 2);
  bf16* wpq = (bf16*)take(16384 * 2);
  bf16* wpk = (bf16*)take(16384 * 2);

  // Decoder small buffers
  bf16*  tlnb   = (bf16*)take(32 * 128 * 2);
  float* qd     = (float*)take(32 * 128 * 4);
  bf16*  headsb = (bf16*)take(32 * 128 * 2);
  float* dh1    = (float*)take(32 * 128 * 4);
  bf16*  dln2b  = (bf16*)take(32 * 128 * 2);
  bf16*  dmid   = (bf16*)take(32 * 512 * 2);
  float* dout   = (float*)take(32 * 128 * 4);
  bf16*  doutb  = (bf16*)take(32 * 128 * 2);
  float* qp     = (float*)take(32 * 128 * 4);

  // ---- Weight prep (transposed bf16) ----
  pack_qkvT_kernel<<<64, 256, 0, stream>>>(enc_wq, wqe);
  pack_qkvT_kernel<<<64, 256, 0, stream>>>(enc_wk, wke);
  pack_qkvT_kernel<<<64, 256, 0, stream>>>(enc_wv, wve);
  pack_qkvT_kernel<<<64, 256, 0, stream>>>(dec_wq, wqd);
  pack_qkvT_kernel<<<64, 256, 0, stream>>>(dec_wk, wkd);
  pack_qkvT_kernel<<<64, 256, 0, stream>>>(dec_wv, wvd);
  castT_kernel<<<64, 256, 0, stream>>>(enc_wo, woe, 128, 128);
  castT_kernel<<<256, 256, 0, stream>>>(enc_w1, w1e, 128, 512);
  castT_kernel<<<256, 256, 0, stream>>>(enc_w2, w2e, 512, 128);
  castT_kernel<<<64, 256, 0, stream>>>(dec_wo, wod, 128, 128);
  castT_kernel<<<256, 256, 0, stream>>>(dec_w1, w1d, 128, 512);
  castT_kernel<<<256, 256, 0, stream>>>(dec_w2, w2d, 512, 128);
  castT_kernel<<<64, 256, 0, stream>>>(ptr_wq, wpq, 128, 128);
  castT_kernel<<<64, 256, 0, stream>>>(ptr_wk, wpk, 128, 128);

  const dim3 g128(2, Mtok / 16);  // N=128 -> 2 x 64-wide blocks
  const dim3 gffn(8, Mtok / 16);  // N=512 -> 8 x 64-wide blocks

  // ---- Encoder ----
  layernorm_kernel<<<Mtok, 32, 0, stream>>>(src, enc_ln1_g, enc_ln1_b, hn);
  gemm_wmma_kernel<<<g128, 32, 0, stream>>>(hn, 128, wqe, 128, nullptr, Qh, nullptr, 128,
                                            128, nullptr, nullptr, 0);
  gemm_wmma_kernel<<<g128, 32, 0, stream>>>(hn, 128, wke, 128, nullptr, Kh, nullptr, 128,
                                            128, nullptr, nullptr, 0);
  gemm_wmma_kernel<<<g128, 32, 0, stream>>>(hn, 128, wve, 128, nullptr, nullptr, Vh, 128,
                                            128, nullptr, nullptr, 0);  // transposed V
  enc_attn_kernel<<<dim3(64, 4, 32), 32, 16 * 1024 * 4, stream>>>(Qh, Kh, Vh, enc_mask, attn);
  gemm_wmma_kernel<<<g128, 32, 0, stream>>>(attn, 128, woe, 128, h1, nullptr, nullptr, 128,
                                            128, nullptr, src, 0);
  layernorm_kernel<<<Mtok, 32, 0, stream>>>(h1, enc_ln2_g, enc_ln2_b, ln2b);
  gemm_wmma_kernel<<<gffn, 32, 0, stream>>>(ln2b, 128, w1e, 128, nullptr, mid, nullptr, 512,
                                            128, enc_b1, nullptr, 1);
  gemm_wmma_kernel<<<g128, 32, 0, stream>>>(mid, 512, w2e, 512, mem, nullptr, nullptr, 128,
                                            512, enc_b2, h1, 0);

  // ---- Decoder (Q=1) ----
  layernorm_kernel<<<32, 32, 0, stream>>>(tgt, dec_ln1_g, dec_ln1_b, tlnb);
  layernorm_kernel<<<Mtok, 32, 0, stream>>>(mem, dec_ln1_g, dec_ln1_b, hn);  // mem_ln
  gemm_wmma_kernel<<<g128, 32, 0, stream>>>(hn, 128, wkd, 128, nullptr, Kh, nullptr, 128,
                                            128, nullptr, nullptr, 0);
  gemm_wmma_kernel<<<g128, 32, 0, stream>>>(hn, 128, wvd, 128, nullptr, nullptr, Vh, 128,
                                            128, nullptr, nullptr, 0);  // transposed V
  small_mm_kernel<<<16, 256, 0, stream>>>(tlnb, 128, wqd, 128, qd, nullptr, 128,
                                          32, 128, 128, nullptr, nullptr, 0);
  dec_attn_kernel<<<dim3(32, 4), 256, 0, stream>>>(qd, Kh, Vh, dec_mask, headsb);
  small_mm_kernel<<<16, 256, 0, stream>>>(headsb, 128, wod, 128, dh1, nullptr, 128,
                                          32, 128, 128, nullptr, tgt, 0);
  layernorm_kernel<<<32, 32, 0, stream>>>(dh1, dec_ln2_g, dec_ln2_b, dln2b);
  small_mm_kernel<<<64, 256, 0, stream>>>(dln2b, 128, w1d, 128, nullptr, dmid, 512,
                                          32, 512, 128, dec_b1, nullptr, 1);
  small_mm_kernel<<<16, 256, 0, stream>>>(dmid, 512, w2d, 512, dout, nullptr, 128,
                                          32, 128, 512, dec_b2, dh1, 0);

  // ---- Pointer head ----
  cast_kernel<<<16, 256, 0, stream>>>(dout, doutb, 32 * 128);
  small_mm_kernel<<<16, 256, 0, stream>>>(doutb, 128, wpq, 128, qp, nullptr, 128,
                                          32, 128, 128, nullptr, nullptr, 0);
  cast_kernel<<<(Mtok * 128 + 255) / 256, 256, 0, stream>>>(mem, ln2b, Mtok * 128);
  gemm_wmma_kernel<<<g128, 32, 0, stream>>>(ln2b, 128, wpk, 128, nullptr, attn, nullptr, 128,
                                            128, nullptr, nullptr, 0);
  pointer_kernel<<<32, 256, 0, stream>>>(qp, attn, dec_mask, (float*)d_out);
}
